// MultiHeadAttention_19430432047180
// MI455X (gfx1250) — matrix operand order repaired and checked by reading, not compiled, not run on hardware
//
#include <hip/hip_runtime.h>
#include <hip/hip_bf16.h>
#include <math.h>

// ---------------- problem constants ----------------
#define HID   1024
#define NH    16
#define HD    64
#define BATCH 2
#define SEQ   2048
#define MTOT  (BATCH * SEQ)          // 4096 rows
#define SCALE 0.125f                 // 1/sqrt(64)

typedef _Float16 v16h  __attribute__((ext_vector_type(16)));
typedef _Float16 half8 __attribute__((ext_vector_type(8)));
typedef float    v8f   __attribute__((ext_vector_type(8)));

// ---------------- helpers ----------------
__device__ __forceinline__ v8f wmma16(v16h a, v16h b, v8f c) {
  return __builtin_amdgcn_wmma_f32_16x16x32_f16(false, a, false, b, (short)0, c,
                                                false, false);
}

// halves p[0..7], p[16..23] -> v16h  (B-operand pattern, table k order; caller adds 8*half)
__device__ __forceinline__ v16h ld16(const _Float16* p) {
  half8 lo = *(const half8*)p;
  half8 hi = *(const half8*)(p + 16);
  return __builtin_shufflevector(lo, hi, 0, 1, 2, 3, 4, 5, 6, 7, 8, 9, 10, 11,
                                 12, 13, 14, 15);
}

// A-operand pattern from an f16 row: a[e] = row[d0 + 8*half + e] (e<8),
//                                    a[8+e] = row[d0 + 16 + 8*half + e]
__device__ __forceinline__ v16h ldA(const _Float16* row, int d0, int half) {
  half8 lo = *(const half8*)(row + d0 + half * 8);
  half8 hi = *(const half8*)(row + d0 + 16 + half * 8);
  return __builtin_shufflevector(lo, hi, 0, 1, 2, 3, 4, 5, 6, 7, 8, 9, 10, 11,
                                 12, 13, 14, 15);
}

#define ZERO8 {0.f, 0.f, 0.f, 0.f, 0.f, 0.f, 0.f, 0.f}

// ---------------- kernel 1: f32 -> f16 convert (weights + x) ----------------
__global__ __launch_bounds__(256) void cvt_kernel(
    const float* __restrict__ qw, const float* __restrict__ kw,
    const float* __restrict__ vw, const float* __restrict__ ow,
    const float* __restrict__ x, _Float16* __restrict__ w_dst,
    _Float16* __restrict__ x_dst) {
  int idx = blockIdx.x * 256 + threadIdx.x;       // 8 * 1M threads
  if (idx < 4 * 1048576) {
    int which = idx >> 20;
    int rem = idx & 1048575;
    float v = (which == 0)   ? qw[rem]
              : (which == 1) ? kw[rem]
              : (which == 2) ? vw[rem]
                             : ow[rem];
    w_dst[idx] = (_Float16)v;
  } else {
    int r = idx - 4 * 1048576;                    // 0 .. 4M-1  (B*S*HID)
    x_dst[r] = (_Float16)x[r];
  }
}

// ---------------- kernel 2: fused QKV projection (WMMA) ----------------
// out[m][n] = sum_k x[m,k] * W[n,k] + bias[n]
// Each wave computes a 16x64 tile (one head width): A reused across 4 WMMAs.
// q,k stored [B,H,S,D] f16 ; v stored transposed [B,H,D,S] f16
__global__ __launch_bounds__(256) void qkv_gemm_kernel(
    const _Float16* __restrict__ x_h, const _Float16* __restrict__ wh,
    const float* __restrict__ qb, const float* __restrict__ kb,
    const float* __restrict__ vb, _Float16* __restrict__ q_ws,
    _Float16* __restrict__ k_ws, _Float16* __restrict__ v_t) {
  int wave = threadIdx.x >> 5;
  int lane = threadIdx.x & 31;
  int ln = lane & 15;
  int half = lane >> 4;

  int gtile = blockIdx.x * 8 + wave;          // 3 * 256 * 16 = 12288 tiles
  int which = gtile / (256 * 16);
  int t = gtile % (256 * 16);
  int m0 = (t / 16) * 16;
  int n0 = (t % 16) * 64;                     // one full head of columns

  const _Float16* W = wh + (size_t)which * (1024 * 1024);
  const float* bias = (which == 0) ? qb : (which == 1) ? kb : vb;

  const _Float16* xrow = x_h + (size_t)(m0 + ln) * HID;            // A (M=ln)
  const _Float16* wr0 = W + (size_t)(n0 + 0 * 16 + ln) * HID;      // B cols
  const _Float16* wr1 = W + (size_t)(n0 + 1 * 16 + ln) * HID;
  const _Float16* wr2 = W + (size_t)(n0 + 2 * 16 + ln) * HID;
  const _Float16* wr3 = W + (size_t)(n0 + 3 * 16 + ln) * HID;

  v8f c0 = ZERO8, c1 = ZERO8, c2 = ZERO8, c3 = ZERO8;
  for (int k0 = 0; k0 < HID; k0 += 32) {
    __builtin_prefetch(wr0 + k0 + 256, 0, 1);
    v16h a = ldA(xrow, k0, half);
    int ko = k0 + half * 8;
    c0 = wmma16(a, ld16(wr0 + ko), c0);
    c1 = wmma16(a, ld16(wr1 + ko), c1);
    c2 = wmma16(a, ld16(wr2 + ko), c2);
    c3 = wmma16(a, ld16(wr3 + ko), c3);
  }

  int h = n0 >> 6;                 // head (tile spans exactly one head)
#pragma unroll
  for (int j = 0; j < 4; ++j) {
    v8f c = (j == 0) ? c0 : (j == 1) ? c1 : (j == 2) ? c2 : c3;
    int d = j * 16 + ln;           // dim within head
    float bv = bias[n0 + d];
#pragma unroll
    for (int r = 0; r < 8; ++r) {
      int mg = m0 + r + 8 * half;  // global row
      int bb = mg >> 11;           // batch
      int s = mg & 2047;           // seq
      _Float16 hv = (_Float16)(c[r] + bv);
      size_t bh = (size_t)(bb * NH + h);
      if (which == 0)
        q_ws[(bh * SEQ + s) * HD + d] = hv;
      else if (which == 1)
        k_ws[(bh * SEQ + s) * HD + d] = hv;
      else
        v_t[(bh * HD + d) * SEQ + s] = hv;
    }
  }
}

// ---------------- kernel 3: RoPE on q,k ----------------
__global__ __launch_bounds__(256) void rope_kernel(_Float16* __restrict__ q_ws,
                                                   _Float16* __restrict__ k_ws) {
  int idx = blockIdx.x * 256 + threadIdx.x;  // B*H*S*32 = 2,097,152 threads
  int j = idx & 31;
  int s = (idx >> 5) & 2047;
  int bh = idx >> 16;                        // b*16 + h  (0..31)
  size_t base = ((size_t)bh * SEQ + s) * HD;

  float inv = powf(10000.f, -(float)j * (1.f / 32.f));
  float ang = (float)s * inv;
  float sn, cs;
  sincosf(ang, &sn, &cs);

  float q1 = (float)q_ws[base + j], q2 = (float)q_ws[base + j + 32];
  q_ws[base + j] = (_Float16)(q1 * cs - q2 * sn);
  q_ws[base + j + 32] = (_Float16)(q2 * cs + q1 * sn);

  float k1 = (float)k_ws[base + j], k2 = (float)k_ws[base + j + 32];
  k_ws[base + j] = (_Float16)(k1 * cs - k2 * sn);
  k_ws[base + j + 32] = (_Float16)(k2 * cs + k1 * sn);
}

// ---------------- kernel 4: flash attention ----------------
// One wave per 16-row q tile.  Scores computed transposed (S^T = K_tile @ Q^T)
// so the softmaxed C-layout maps directly onto the A-layout of the P@V WMMA.
__global__ __launch_bounds__(256) void attn_kernel(
    const _Float16* __restrict__ q_ws, const _Float16* __restrict__ k_ws,
    const _Float16* __restrict__ v_t, _Float16* __restrict__ attn_out) {
  int wave = threadIdx.x >> 5;
  int lane = threadIdx.x & 31;
  int n = lane & 15;
  int half = lane >> 4;

  int qt = blockIdx.x * 8 + wave;        // 0..4095
  int s0 = (qt & 127) << 4;
  int h = (qt >> 7) & 15;
  int bb = qt >> 11;

  size_t bh = (size_t)(bb * NH + h);
  const _Float16* qbase = q_ws + bh * SEQ * HD;
  const _Float16* kbase = k_ws + bh * SEQ * HD;
  const _Float16* vbase = v_t + bh * HD * SEQ;

  // B operand from Q (column n = q-row s0+n), held for the whole loop
  const _Float16* qrow = qbase + (size_t)(s0 + n) * HD;
  v16h bq0 = ld16(qrow + half * 8);          // d = 0..31
  v16h bq1 = ld16(qrow + 32 + half * 8);     // d = 32..63

  v8f acc0 = ZERO8, acc1 = ZERO8, acc2 = ZERO8, acc3 = ZERO8;
  float mrow = -INFINITY, lrow = 0.f;

  for (int kt = 0; kt < SEQ; kt += 32) {
    // A operands: 32 keys (lane's A-row m == n == key-in-tile)
    const _Float16* krow0 = kbase + (size_t)(kt + n) * HD;
    const _Float16* krow1 = kbase + (size_t)(kt + 16 + n) * HD;
    v8f c0 = ZERO8, c1 = ZERO8;
    c0 = wmma16(ldA(krow0, 0, half), bq0, c0);
    c0 = wmma16(ldA(krow0, 32, half), bq1, c0);
    c1 = wmma16(ldA(krow1, 0, half), bq0, c1);
    c1 = wmma16(ldA(krow1, 32, half), bq1, c1);

    // lane holds, for q-row (s0+n): keys kt+8*half+r (c0) and kt+16+8*half+r (c1)
    float mx = -INFINITY;
#pragma unroll
    for (int r = 0; r < 8; ++r) {
      c0[r] *= SCALE;
      c1[r] *= SCALE;
      mx = fmaxf(mx, fmaxf(c0[r], c1[r]));
    }
    mx = fmaxf(mx, __shfl_xor(mx, 16, 32));        // combine key-halves
    float mnew = fmaxf(mrow, mx);
    float ef = __expf(mrow - mnew);                // 0 on first iter

    float rs = 0.f;
    v16h ap;                                       // P in A-layout, directly
#pragma unroll
    for (int r = 0; r < 8; ++r) {
      float p0 = __expf(c0[r] - mnew);
      float p1 = __expf(c1[r] - mnew);
      rs += p0 + p1;
      ap[r] = (_Float16)p0;
      ap[8 + r] = (_Float16)p1;
    }
    rs += __shfl_xor(rs, 16, 32);
    lrow = lrow * ef + rs;
    mrow = mnew;

    // rescale accumulators: out-row of element r is (r + 8*half); its factor
    // lives in the lane whose (lane&15) == that row.
#pragma unroll
    for (int r = 0; r < 8; ++r) {
      float f = __shfl(ef, r + 8 * half, 32);
      acc0[r] *= f;
      acc1[r] *= f;
      acc2[r] *= f;
      acc3[r] *= f;
    }

    // P @ V : B operand columns are head-dims, rows are keys (contiguous in
    // the transposed V layout)
    const _Float16* vcol = vbase + kt + half * 8;
    v16h bv0 = ld16(vcol + (size_t)(0 * 16 + n) * SEQ);
    v16h bv1 = ld16(vcol + (size_t)(1 * 16 + n) * SEQ);
    v16h bv2 = ld16(vcol + (size_t)(2 * 16 + n) * SEQ);
    v16h bv3 = ld16(vcol + (size_t)(3 * 16 + n) * SEQ);
    acc0 = wmma16(ap, bv0, acc0);
    acc1 = wmma16(ap, bv1, acc1);
    acc2 = wmma16(ap, bv2, acc2);
    acc3 = wmma16(ap, bv3, acc3);
  }

  // normalize and store as f16 [B, S, H*D]
#pragma unroll
  for (int r = 0; r < 8; ++r) {
    float linv = 1.f / __shfl(lrow, r + 8 * half, 32);
    int s = s0 + r + 8 * half;
    _Float16* orow = attn_out + ((size_t)(bb * SEQ + s)) * HID + h * HD;
    orow[0 * 16 + n] = (_Float16)(acc0[r] * linv);
    orow[1 * 16 + n] = (_Float16)(acc1[r] * linv);
    orow[2 * 16 + n] = (_Float16)(acc2[r] * linv);
    orow[3 * 16 + n] = (_Float16)(acc3[r] * linv);
  }
}

// ---------------- kernel 5: output projection (WMMA, 16x64 tiles) ----------
__global__ __launch_bounds__(256) void oproj_kernel(
    const _Float16* __restrict__ attn, const _Float16* __restrict__ ow_h,
    const float* __restrict__ ob, float* __restrict__ out) {
  int wave = threadIdx.x >> 5;
  int lane = threadIdx.x & 31;
  int ln = lane & 15;
  int half = lane >> 4;

  int gtile = blockIdx.x * 8 + wave;    // 256 * 16 = 4096 tiles
  int m0 = (gtile / 16) * 16;
  int n0 = (gtile % 16) * 64;

  const _Float16* arow = attn + (size_t)(m0 + ln) * HID;
  const _Float16* wr0 = ow_h + (size_t)(n0 + 0 * 16 + ln) * HID;
  const _Float16* wr1 = ow_h + (size_t)(n0 + 1 * 16 + ln) * HID;
  const _Float16* wr2 = ow_h + (size_t)(n0 + 2 * 16 + ln) * HID;
  const _Float16* wr3 = ow_h + (size_t)(n0 + 3 * 16 + ln) * HID;

  v8f c0 = ZERO8, c1 = ZERO8, c2 = ZERO8, c3 = ZERO8;
  for (int k0 = 0; k0 < HID; k0 += 32) {
    __builtin_prefetch(wr0 + k0 + 256, 0, 1);
    v16h a = ldA(arow, k0, half);
    int ko = k0 + half * 8;
    c0 = wmma16(a, ld16(wr0 + ko), c0);
    c1 = wmma16(a, ld16(wr1 + ko), c1);
    c2 = wmma16(a, ld16(wr2 + ko), c2);
    c3 = wmma16(a, ld16(wr3 + ko), c3);
  }

#pragma unroll
  for (int j = 0; j < 4; ++j) {
    v8f c = (j == 0) ? c0 : (j == 1) ? c1 : (j == 2) ? c2 : c3;
    float bv = ob[n0 + j * 16 + ln];
#pragma unroll
    for (int r = 0; r < 8; ++r)
      out[(size_t)(m0 + r + 8 * half) * HID + n0 + j * 16 + ln] = c[r] + bv;
  }
}

// ---------------- launch ----------------
extern "C" void kernel_launch(void* const* d_in, const int* in_sizes, int n_in,
                              void* d_out, int out_size, void* d_ws,
                              size_t ws_size, hipStream_t stream) {
  (void)in_sizes; (void)n_in; (void)out_size; (void)ws_size;
  const float* x  = (const float*)d_in[0];
  const float* qw = (const float*)d_in[1];
  const float* qb = (const float*)d_in[2];
  const float* kw = (const float*)d_in[3];
  const float* kb = (const float*)d_in[4];
  const float* vw = (const float*)d_in[5];
  const float* vb = (const float*)d_in[6];
  const float* ow = (const float*)d_in[7];
  const float* ob = (const float*)d_in[8];
  float* out = (float*)d_out;

  _Float16* ws = (_Float16*)d_ws;
  const size_t M1 = 1048576;
  _Float16* w_h  = ws;              // 4 * 1M f16 (qw|kw|vw|ow)
  _Float16* x_h  = ws + 4 * M1;     // [B*S, HID] f16
  _Float16* q_ws = ws + 8 * M1;     // [B,H,S,D]
  _Float16* k_ws = ws + 12 * M1;    // [B,H,S,D]
  _Float16* v_t  = ws + 16 * M1;    // [B,H,D,S]
  _Float16* attn = ws + 20 * M1;    // [B,S,HID]   total 48 MB

  cvt_kernel<<<32768, 256, 0, stream>>>(qw, kw, vw, ow, x, w_h, x_h);
  qkv_gemm_kernel<<<1536, 256, 0, stream>>>(x_h, w_h, qb, kb, vb, q_ws, k_ws,
                                            v_t);
  rope_kernel<<<8192, 256, 0, stream>>>(q_ws, k_ws);
  attn_kernel<<<512, 256, 0, stream>>>(q_ws, k_ws, v_t, attn);
  oproj_kernel<<<512, 256, 0, stream>>>(attn, w_h + 3 * M1, ob, out);
}
